// BlockSparseMLP_32899449487637
// MI455X (gfx1250) — matrix operand order repaired and checked by reading, not compiled, not run on hardware
//
#include <hip/hip_runtime.h>
#include <hip/hip_bf16.h>

// MoE gated MLP for MI455X (gfx1250): top-2 routed grouped GEMM in bf16 WMMA.
// T=2048 tokens, H=2048 hidden, I=4096 intermediate, E=8 experts, K=2.

#define T_TOK 2048
#define H_DIM 2048
#define I_DIM 4096
#define N_EXP 8

#define BM 64    // gathered token rows per block (4 x 16 WMMA M-tiles)
#define BN 256   // columns per block (8 waves x 32)
#define KT 32    // K-step (matches 16x16x32 bf16 WMMA)

typedef __attribute__((ext_vector_type(16))) __bf16 v16bf;
typedef __attribute__((ext_vector_type(8)))  float  v8f;

union Frag16 { v16bf v; uint4 q[2]; };

// ---------------------------------------------------------------------------
// Kernel 0: zero output (harness poisons it), zero counters/token lists,
// convert x fp32 -> bf16 for WMMA A operands.
// ---------------------------------------------------------------------------
__global__ void k_init(const float* __restrict__ x, __bf16* __restrict__ xb,
                       float* __restrict__ out, int* __restrict__ cnt,
                       int* __restrict__ tok, float* __restrict__ wt) {
  size_t idx = (size_t)blockIdx.x * blockDim.x + threadIdx.x;
  size_t n = (size_t)T_TOK * H_DIM;
  if (idx < n) { out[idx] = 0.f; xb[idx] = (__bf16)x[idx]; }
  if (idx < N_EXP) cnt[idx] = 0;
  if (idx < (size_t)N_EXP * T_TOK) { tok[idx] = 0; wt[idx] = 0.f; }
}

// ---------------------------------------------------------------------------
// Kernel 1: router. One thread per token: 8 dot products of length H,
// top-2 (lowest-index tie-break), softmax over the pair, scatter into
// per-expert (token, weight) lists via atomic slot counters.
// ---------------------------------------------------------------------------
__global__ void k_router(const float* __restrict__ x, const float* __restrict__ wr,
                         int* __restrict__ cnt, int* __restrict__ tok,
                         float* __restrict__ wt) {
  int t = blockIdx.x * blockDim.x + threadIdx.x;
  if (t >= T_TOK) return;
  float acc[N_EXP];
#pragma unroll
  for (int e = 0; e < N_EXP; ++e) acc[e] = 0.f;
  const float* xr = x + (size_t)t * H_DIM;
  for (int h = 0; h < H_DIM; ++h) {
    float xv = xr[h];
    const float4* w4 = reinterpret_cast<const float4*>(wr + (size_t)h * N_EXP);
    float4 a = w4[0], b = w4[1];
    acc[0] += xv * a.x; acc[1] += xv * a.y; acc[2] += xv * a.z; acc[3] += xv * a.w;
    acc[4] += xv * b.x; acc[5] += xv * b.y; acc[6] += xv * b.z; acc[7] += xv * b.w;
  }
  int e0 = 0; float v0 = acc[0];
#pragma unroll
  for (int e = 1; e < N_EXP; ++e) if (acc[e] > v0) { v0 = acc[e]; e0 = e; }
  int e1 = (e0 == 0) ? 1 : 0; float v1 = acc[e1];
#pragma unroll
  for (int e = 0; e < N_EXP; ++e)
    if (e != e0 && acc[e] > v1) { v1 = acc[e]; e1 = e; }
  float s1 = __expf(v1 - v0);                 // v0 >= v1
  float w0 = __builtin_amdgcn_rcpf(1.f + s1); // softmax over the pair
  float w1 = s1 * w0;
  int slot0 = atomicAdd(&cnt[e0], 1);
  tok[e0 * T_TOK + slot0] = t; wt[e0 * T_TOK + slot0] = w0;
  int slot1 = atomicAdd(&cnt[e1], 1);
  tok[e1 * T_TOK + slot1] = t; wt[e1 * T_TOK + slot1] = w1;
}

// ---------------------------------------------------------------------------
// WMMA fragment layouts (CDNA5 ISA §7.12.2, wave32):
//  A (16x32 bf16): lane m=l&15; halves 0..7 = K0..K0+7, 8..15 = K0+16..K0+23,
//                  K0 = (l<16)?0:8  -> two 16B LDS reads from row-major tile.
//  B (32x16 bf16): lane n=l&15; halves 0..7 = K0..K0+7, 8..15 = K0+16..K0+23, K0 = (l<16)?0:8
//                  -> two 16B LDS reads from K-contiguous (transposed) tile.
//  C/D (16x16 f32): vgpr r, lane l: row = r + (l<16?0:8), col = l&15.
// ---------------------------------------------------------------------------

// Kernel 2: fused gate/up grouped GEMM + silu_mul, bf16 activations out.
// grid.x = I/BN, grid.y = E * (T/BM). Blocks past an expert's count exit.
__global__ __launch_bounds__(256)
void k_gateup(const __bf16* __restrict__ xb,
              const float* __restrict__ wg, const float* __restrict__ wu,
              const int* __restrict__ cnt, const int* __restrict__ tok,
              __bf16* __restrict__ act) {
  __shared__ __align__(16) __bf16 sA[BM][KT];
  __shared__ __align__(16) __bf16 sBg[BN][KT];   // transposed: [n][k]
  __shared__ __align__(16) __bf16 sBu[BN][KT];
  __shared__ int sTok[BM];

  int e = blockIdx.y >> 5;            // / (T/BM) = 32
  int rowTile = blockIdx.y & 31;
  int count = cnt[e];
  if (rowTile * BM >= count) return;
  int n0 = blockIdx.x * BN;
  int tid = threadIdx.x;
  int lane = tid & 31, wave = tid >> 5;

  if (tid < BM) sTok[tid] = tok[e * T_TOK + rowTile * BM + tid];
  __syncthreads();

  v8f accG[4][2] = {};
  v8f accU[4][2] = {};
  size_t wbase = (size_t)e * H_DIM * I_DIM;
  int arow = tid >> 2, ach = (tid & 3) * 8;     // A-tile load mapping

  for (int k0 = 0; k0 < H_DIM; k0 += KT) {
    // A tile: 64 gathered rows x 32 halves; one 16B load per thread.
    {
      int tk = sTok[arow];
      uint4 av = *reinterpret_cast<const uint4*>(xb + (size_t)tk * H_DIM + k0 + ach);
      *reinterpret_cast<uint4*>(&sA[arow][ach]) = av;
    }
    // B tiles: 32(K) x 256(N) fp32 -> bf16, transposed into LDS [n][k].
    {
      const float* g = wg + wbase + (size_t)k0 * I_DIM + n0 + tid;
      const float* u = wu + wbase + (size_t)k0 * I_DIM + n0 + tid;
      if (k0 + KT < H_DIM) {   // hint next K-panel into L2 (global_prefetch)
        __builtin_prefetch(g + (size_t)KT * I_DIM, 0, 1);
        __builtin_prefetch(u + (size_t)KT * I_DIM, 0, 1);
      }
#pragma unroll 4
      for (int j = 0; j < KT; ++j) {
        sBg[tid][j] = (__bf16)g[(size_t)j * I_DIM];
        sBu[tid][j] = (__bf16)u[(size_t)j * I_DIM];
      }
    }
    __syncthreads();

    int m = lane & 15;
    int ak = (lane < 16) ? 0 : 8;
    int bk = (lane < 16) ? 0 : 8;
    int nb = wave * 32 + (lane & 15);

    Frag16 a[4], bg[2], bu[2];
#pragma unroll
    for (int mt = 0; mt < 4; ++mt) {
      a[mt].q[0] = *reinterpret_cast<const uint4*>(&sA[mt * 16 + m][ak]);
      a[mt].q[1] = *reinterpret_cast<const uint4*>(&sA[mt * 16 + m][ak + 16]);
    }
#pragma unroll
    for (int ns = 0; ns < 2; ++ns) {
      bg[ns].q[0] = *reinterpret_cast<const uint4*>(&sBg[nb + ns * 16][bk]);
      bg[ns].q[1] = *reinterpret_cast<const uint4*>(&sBg[nb + ns * 16][bk + 16]);
      bu[ns].q[0] = *reinterpret_cast<const uint4*>(&sBu[nb + ns * 16][bk]);
      bu[ns].q[1] = *reinterpret_cast<const uint4*>(&sBu[nb + ns * 16][bk + 16]);
    }
#pragma unroll
    for (int mt = 0; mt < 4; ++mt) {
#pragma unroll
      for (int ns = 0; ns < 2; ++ns) {
        accG[mt][ns] = __builtin_amdgcn_wmma_f32_16x16x32_bf16(
            false, a[mt].v, false, bg[ns].v, (short)0, accG[mt][ns], false, false);
        accU[mt][ns] = __builtin_amdgcn_wmma_f32_16x16x32_bf16(
            false, a[mt].v, false, bu[ns].v, (short)0, accU[mt][ns], false, false);
      }
    }
    __syncthreads();
  }

  // Epilogue: act = silu(g) * u = g * rcp(1+exp(-g)) * u, stored bf16.
  size_t rowGBase = (size_t)(e * T_TOK + rowTile * BM);
  int mHi = (lane >> 4) << 3;
  int colB = n0 + wave * 32 + (lane & 15);
#pragma unroll
  for (int mt = 0; mt < 4; ++mt) {
#pragma unroll
    for (int ns = 0; ns < 2; ++ns) {
      int col = colB + ns * 16;
#pragma unroll
      for (int r = 0; r < 8; ++r) {
        int m = mt * 16 + r + mHi;
        float g = accG[mt][ns][r];
        float s = __builtin_amdgcn_rcpf(1.f + __expf(-g));
        act[(rowGBase + m) * I_DIM + col] = (__bf16)(g * s * accU[mt][ns][r]);
      }
    }
  }
}

// Kernel 3: down-projection grouped GEMM + weighted combine into out.
// grid.x = H/BN, grid.y = E * (T/BM). Exactly 2 atomic adds per out element
// (plus zero init) -> order-independent, deterministic.
__global__ __launch_bounds__(256)
void k_down(const __bf16* __restrict__ act, const float* __restrict__ wd,
            const int* __restrict__ cnt, const int* __restrict__ tok,
            const float* __restrict__ wt, float* __restrict__ out) {
  __shared__ __align__(16) __bf16 sA[BM][KT];
  __shared__ __align__(16) __bf16 sB[BN][KT];
  __shared__ int sTok[BM];
  __shared__ float sW[BM];

  int e = blockIdx.y >> 5;
  int rowTile = blockIdx.y & 31;
  int count = cnt[e];
  if (rowTile * BM >= count) return;
  int n0 = blockIdx.x * BN;
  int tid = threadIdx.x;
  int lane = tid & 31, wave = tid >> 5;

  if (tid < BM) {
    sTok[tid] = tok[e * T_TOK + rowTile * BM + tid];
    sW[tid]   = wt[e * T_TOK + rowTile * BM + tid];
  }
  __syncthreads();

  v8f acc[4][2] = {};
  size_t rowGBase = (size_t)(e * T_TOK + rowTile * BM);
  size_t wbase = (size_t)e * I_DIM * H_DIM;
  int arow = tid >> 2, ach = (tid & 3) * 8;

  for (int k0 = 0; k0 < I_DIM; k0 += KT) {
    {
      uint4 av = *reinterpret_cast<const uint4*>(
          act + (rowGBase + arow) * I_DIM + k0 + ach);
      *reinterpret_cast<uint4*>(&sA[arow][ach]) = av;
    }
    {
      const float* d = wd + wbase + (size_t)k0 * H_DIM + n0 + tid;
      if (k0 + KT < I_DIM)
        __builtin_prefetch(d + (size_t)KT * H_DIM, 0, 1);
#pragma unroll 4
      for (int j = 0; j < KT; ++j) sB[tid][j] = (__bf16)d[(size_t)j * H_DIM];
    }
    __syncthreads();

    int m = lane & 15;
    int ak = (lane < 16) ? 0 : 8;
    int bk = (lane < 16) ? 0 : 8;
    int nb = wave * 32 + (lane & 15);

    Frag16 a[4], b[2];
#pragma unroll
    for (int mt = 0; mt < 4; ++mt) {
      a[mt].q[0] = *reinterpret_cast<const uint4*>(&sA[mt * 16 + m][ak]);
      a[mt].q[1] = *reinterpret_cast<const uint4*>(&sA[mt * 16 + m][ak + 16]);
    }
#pragma unroll
    for (int ns = 0; ns < 2; ++ns) {
      b[ns].q[0] = *reinterpret_cast<const uint4*>(&sB[nb + ns * 16][bk]);
      b[ns].q[1] = *reinterpret_cast<const uint4*>(&sB[nb + ns * 16][bk + 16]);
    }
#pragma unroll
    for (int mt = 0; mt < 4; ++mt)
#pragma unroll
      for (int ns = 0; ns < 2; ++ns)
        acc[mt][ns] = __builtin_amdgcn_wmma_f32_16x16x32_bf16(
            false, a[mt].v, false, b[ns].v, (short)0, acc[mt][ns], false, false);
    __syncthreads();
  }

  int mHi = (lane >> 4) << 3;
  int colB = n0 + wave * 32 + (lane & 15);
#pragma unroll
  for (int mt = 0; mt < 4; ++mt) {
#pragma unroll
    for (int ns = 0; ns < 2; ++ns) {
      int col = colB + ns * 16;
#pragma unroll
      for (int r = 0; r < 8; ++r) {
        int m = mt * 16 + r + mHi;
        if (rowTile * BM + m < count) {
          unsafeAtomicAdd(&out[(size_t)sTok[m] * H_DIM + col],
                          acc[mt][ns][r] * sW[m]);
        }
      }
    }
  }
}

// ---------------------------------------------------------------------------
extern "C" void kernel_launch(void* const* d_in, const int* in_sizes, int n_in,
                              void* d_out, int out_size, void* d_ws, size_t ws_size,
                              hipStream_t stream) {
  const float* x        = (const float*)d_in[0];
  const float* w_router = (const float*)d_in[1];
  const float* w_gate   = (const float*)d_in[2];
  const float* w_up     = (const float*)d_in[3];
  const float* w_down   = (const float*)d_in[4];
  float* out = (float*)d_out;

  // Workspace carve-up (~143 MB total):
  char* ws = (char*)d_ws;
  __bf16* xb = (__bf16*)ws;                 ws += (size_t)T_TOK * H_DIM * 2;
  __bf16* act = (__bf16*)ws;                ws += (size_t)N_EXP * T_TOK * I_DIM * 2;
  int* cnt = (int*)ws;                      ws += 128;
  int* tok = (int*)ws;                      ws += (size_t)N_EXP * T_TOK * 4;
  float* wt = (float*)ws;                   /* end */

  size_t n = (size_t)T_TOK * H_DIM;
  k_init<<<dim3((unsigned)((n + 255) / 256)), dim3(256), 0, stream>>>(x, xb, out, cnt, tok, wt);
  k_router<<<dim3(T_TOK / 256), dim3(256), 0, stream>>>(x, w_router, cnt, tok, wt);
  dim3 g1(I_DIM / BN, N_EXP * (T_TOK / BM));
  k_gateup<<<g1, dim3(256), 0, stream>>>(xb, w_gate, w_up, cnt, tok, act);
  dim3 g2(H_DIM / BN, N_EXP * (T_TOK / BM));
  k_down<<<g2, dim3(256), 0, stream>>>(act, w_down, cnt, tok, wt, out);
}
